// MultiHeadSelfAttention_65317862638411
// MI455X (gfx1250) — matrix operand order repaired and checked by reading, not compiled, not run on hardware
//
#include <hip/hip_runtime.h>
#include <hip/hip_bf16.h>

// MI455X / gfx1250, wave32. All GEMMs via v_wmma_f32_16x16x32_bf16.
// Pipeline: f32->bf16 convert; QKV projection + RoPE (fused); flash attention
// (online softmax, scores never leave the WGP); output projection.

typedef __attribute__((ext_vector_type(8)))  float  v8f;
typedef __attribute__((ext_vector_type(8)))  __bf16 v8bf;
typedef __attribute__((ext_vector_type(16))) __bf16 v16bf;

#define B_   4
#define S_   2048
#define D_   1024
#define H_   16
#define DK_  64
#define NROW (B_ * S_)   // 8192

__device__ __forceinline__ void lds_fence() {
    // wave-local LDS ordering: DS ops are in-order per wave; waiting DScnt==0
    // makes cross-lane LDS writes visible to subsequent reads in this wave.
    asm volatile("s_wait_dscnt 0" ::: "memory");
}

__device__ __forceinline__ v16bf cat8(v8bf lo, v8bf hi) {
    return __builtin_shufflevector(lo, hi, 0,1,2,3,4,5,6,7,8,9,10,11,12,13,14,15);
}

// A fragment (16xK=32 bf16), row-major source, row stride ld (elements).
// ISA layout: lanes 0-15 hold row M=lane, K={0..7,16..23}; lanes 16-31 same
// rows, K={8..15,24..31}.
__device__ __forceinline__ v16bf loadA(const __bf16* __restrict__ src, size_t ld,
                                       int row0, int k0, int lane) {
    int r  = lane & 15;
    int hb = (lane >> 4) & 1;
    const __bf16* p = src + (size_t)(row0 + r) * ld + k0 + hb * 8;
    v8bf lo = *(const v8bf*)p;
    v8bf hi = *(const v8bf*)(p + 16);
    return cat8(lo, hi);
}

// B fragment (K=32 x 16 bf16). Source laid out so each output column n is a
// contiguous K-run: src + n*ldn + k. Lanes 0-15: K={0..7,16..23}, lanes 16-31:
// K={8..15,24..31} (relative to k0), column n = n0 + (lane&15).
__device__ __forceinline__ v16bf loadB(const __bf16* __restrict__ src, size_t ldn,
                                       int n0, int k0, int lane) {
    int n  = n0 + (lane & 15);
    int hb = (lane >> 4) & 1;
    const __bf16* p = src + (size_t)n * ldn + k0 + hb * 8;
    return cat8(*(const v8bf*)p, *(const v8bf*)(p + 16));
}

__device__ __forceinline__ v8f wmma_bf16(v16bf a, v16bf b, v8f c) {
    return __builtin_amdgcn_wmma_f32_16x16x32_bf16(false, a, false, b,
                                                   (short)0, c, false, false);
}

// ---------------------------------------------------------------- convert
__global__ void f2bf_kernel(const float* __restrict__ src,
                            __bf16* __restrict__ dst, int n) {
    int i = (blockIdx.x * blockDim.x + threadIdx.x) * 4;
    if (i + 3 < n) {
        float4 v = *(const float4*)(src + i);
        dst[i + 0] = (__bf16)v.x;
        dst[i + 1] = (__bf16)v.y;
        dst[i + 2] = (__bf16)v.z;
        dst[i + 3] = (__bf16)v.w;
    }
}

// ------------------------------------------------- QKV projection (+RoPE)
// out[n, o] = sum_d x[n,d] * W[o,d].  Each wave: 16 rows x 32 cols.
// which: 0=Q (RoPE + 1/8 scale, out [b,h,s,64]), 1=K (RoPE, out [b,h,s,64]),
//        2=V (no RoPE, out transposed [b,h,64,S]).
__global__ void qkv_kernel(const __bf16* __restrict__ xb,
                           const __bf16* __restrict__ wb,
                           const int* __restrict__ pos,
                           __bf16* __restrict__ out, int which) {
    __shared__ float s_tile[8][16][16];
    const int lane = threadIdx.x & 31;
    const int wave = threadIdx.x >> 5;
    const int n0   = (blockIdx.x * 8 + wave) * 32;
    const int row0 = blockIdx.y * 16;

    v8f c0 = {}, c1 = {};
    for (int k0 = 0; k0 < D_; k0 += 32) {
        v16bf a  = loadA(xb, D_, row0, k0, lane);
        v16bf b0 = loadB(wb, D_, n0,      k0, lane);
        v16bf b1 = loadB(wb, D_, n0 + 16, k0, lane);
        c0 = wmma_bf16(a, b0, c0);
        c1 = wmma_bf16(a, b1, c1);
    }

    const int hb  = (lane >> 4) & 1;
    const int col = lane & 15;
    for (int f = 0; f < 2; ++f) {
        v8f c  = f ? c1 : c0;
        int nb = n0 + f * 16;
        if (which == 2) {
            // V: transposed store [b,h,dk,S]; 8 consecutive tokens per lane.
            int ch = nb + col;
            int hh = ch >> 6, dk = ch & 63;
            int b  = row0 >> 11;
            int s  = (row0 & (S_ - 1)) + hb * 8;
            v8bf pv;
            for (int r = 0; r < 8; ++r) pv[r] = (__bf16)c[r];
            __bf16* dst = out + ((size_t)(b * H_ + hh) * DK_ + dk) * S_ + s;
            *(v8bf*)dst = pv;
        } else {
            // stage C tile, re-map lanes to (row, even/odd pair) for RoPE
            for (int r = 0; r < 8; ++r)
                s_tile[wave][r + hb * 8][col] = c[r];
            lds_fence();
            const float scaleq = (which == 0) ? 0.125f : 1.0f; // 1/sqrt(64)
            for (int t = 0; t < 4; ++t) {
                int pr  = lane + 32 * t;       // 128 pairs: 16 rows x 8 pairs
                int row = pr >> 3;
                int pj  = pr & 7;
                float x1 = s_tile[wave][row][2 * pj];
                float x2 = s_tile[wave][row][2 * pj + 1];
                int sg = row0 + row;
                int b  = sg >> 11;
                int s  = sg & (S_ - 1);
                int ch = nb + 2 * pj;
                int hh = ch >> 6;
                int dk0 = ch & 63;
                int i   = dk0 >> 1;
                // theta^(-i/32) = exp(-i * ln(10000)/32)
                float inv = __expf((float)i * -0.2878231366242146f);
                float ang = (float)pos[sg] * inv;
                float sn, cs;
                __sincosf(ang, &sn, &cs);
                float o1 = (x1 * cs - x2 * sn) * scaleq;
                float o2 = (x1 * sn + x2 * cs) * scaleq;
                size_t idx = ((size_t)(b * H_ + hh) * S_ + s) * DK_ + dk0;
                out[idx]     = (__bf16)o1;
                out[idx + 1] = (__bf16)o2;
            }
            lds_fence();
        }
    }
}

// --------------------------------------------------------- flash attention
// One wave per 16-query tile; key loop in steps of 32 up to causal bound.
__global__ void attn_kernel(const __bf16* __restrict__ Qb,
                            const __bf16* __restrict__ Kb,
                            const __bf16* __restrict__ Vt,
                            __bf16* __restrict__ attn_out) {
    __shared__ float  s_sc[8][16][32];
    __shared__ __bf16 s_P [8][16][32];
    __shared__ float  s_fac[8][16];

    const int lane = threadIdx.x & 31;
    const int wave = threadIdx.x >> 5;
    const int bh   = blockIdx.y;            // b*16 + h
    const int b    = bh >> 4;
    const int h    = bh & 15;
    const int q0   = (blockIdx.x * 8 + wave) * 16;

    const __bf16* Qp = Qb + (size_t)bh * S_ * DK_;
    const __bf16* Kp = Kb + (size_t)bh * S_ * DK_;
    const __bf16* Vp = Vt + (size_t)bh * DK_ * S_;

    v16bf aq0 = loadA(Qp, DK_, q0, 0,  lane);
    v16bf aq1 = loadA(Qp, DK_, q0, 32, lane);

    v8f o[4] = {{}, {}, {}, {}};
    float m_r = -__builtin_inff();
    float l_r = 0.0f;

    const int hb  = (lane >> 4) & 1;
    const int col = lane & 15;

    for (int k0 = 0; k0 <= q0 + 15; k0 += 32) {
        // scores S[16q x 32k] = Q(16x64) K^T
        for (int f = 0; f < 2; ++f) {
            v8f sc = {};
            v16bf bk0 = loadB(Kp, DK_, k0 + f * 16, 0,  lane);
            sc = wmma_bf16(aq0, bk0, sc);
            v16bf bk1 = loadB(Kp, DK_, k0 + f * 16, 32, lane);
            sc = wmma_bf16(aq1, bk1, sc);
            for (int r = 0; r < 8; ++r)
                s_sc[wave][r + hb * 8][f * 16 + col] = sc[r];
        }
        lds_fence();

        if (lane < 16) {              // lanes 0-15 own one query row each
            int row  = lane;
            int jmax = q0 + row - k0; // causal: key k0+j valid iff j<=jmax
            float mold = m_r;
            float tmax = mold;
            for (int j = 0; j < 32; ++j)
                if (j <= jmax) tmax = fmaxf(tmax, s_sc[wave][row][j]);
            float scale = __expf(mold - tmax);   // exp(-inf)=0 on first tile
            float sum = 0.0f;
            for (int j = 0; j < 32; ++j) {
                float p = (j <= jmax) ? __expf(s_sc[wave][row][j] - tmax) : 0.0f;
                sum += p;
                s_P[wave][row][j] = (__bf16)p;
            }
            m_r = tmax;
            l_r = l_r * scale + sum;
            s_fac[wave][row] = scale;
        }
        lds_fence();

        // rescale running O by per-row factor (C-layout row = r + hb*8)
        float fr[8];
        for (int r = 0; r < 8; ++r) fr[r] = s_fac[wave][hb * 8 + r];
        for (int nt = 0; nt < 4; ++nt)
            for (int r = 0; r < 8; ++r) o[nt][r] *= fr[r];

        // reload P from LDS in A-operand layout
        const __bf16* pp = &s_P[wave][col][hb * 8];
        v8bf plo = *(const v8bf*)pp;
        v8bf phi = *(const v8bf*)(pp + 16);
        v16bf pa = cat8(plo, phi);

        // O[16x64] += P(16x32) * V(32x64); V^T layout gives contiguous B runs
        for (int nt = 0; nt < 4; ++nt) {
            v16bf vb = loadB(Vp, S_, nt * 16, k0, lane);
            o[nt] = wmma_bf16(pa, vb, o[nt]);
        }
        lds_fence();   // before next iter overwrites s_sc / s_P
    }

    if (lane < 16) s_fac[wave][lane] = 1.0f / l_r;
    lds_fence();
    {
        float fr[8];
        for (int r = 0; r < 8; ++r) fr[r] = s_fac[wave][hb * 8 + r];
        // stage final tile as bf16 [16][64] in (reused) s_sc, then coalesce
        __bf16* ob = (__bf16*)&s_sc[wave][0][0];
        for (int nt = 0; nt < 4; ++nt)
            for (int r = 0; r < 8; ++r)
                ob[(r + hb * 8) * 64 + nt * 16 + col] = (__bf16)(o[nt][r] * fr[r]);
        lds_fence();
        int row  = lane >> 1;
        int half = (lane & 1) * 32;   // 32 bf16 = 64B per lane
        const uint4* src = (const uint4*)&ob[row * 64 + half];
        uint4* dst = (uint4*)(attn_out +
                     ((size_t)(b * S_ + q0 + row)) * D_ + h * DK_ + half);
        dst[0] = src[0]; dst[1] = src[1]; dst[2] = src[2]; dst[3] = src[3];
    }
}

// -------------------------------------------------------- output projection
// out[n, d] = sum_o attn[n,o] * Wo[d,o]  (fp32 result)
__global__ void proj_kernel(const __bf16* __restrict__ ab,
                            const __bf16* __restrict__ wo,
                            float* __restrict__ out) {
    const int lane = threadIdx.x & 31;
    const int wave = threadIdx.x >> 5;
    const int n0   = (blockIdx.x * 8 + wave) * 32;
    const int row0 = blockIdx.y * 16;

    v8f c0 = {}, c1 = {};
    for (int k0 = 0; k0 < D_; k0 += 32) {
        v16bf a  = loadA(ab, D_, row0, k0, lane);
        v16bf b0 = loadB(wo, D_, n0,      k0, lane);
        v16bf b1 = loadB(wo, D_, n0 + 16, k0, lane);
        c0 = wmma_bf16(a, b0, c0);
        c1 = wmma_bf16(a, b1, c1);
    }
    const int hb  = (lane >> 4) & 1;
    const int col = lane & 15;
    for (int f = 0; f < 2; ++f) {
        v8f c = f ? c1 : c0;
        int n = n0 + f * 16 + col;
        for (int r = 0; r < 8; ++r)
            out[(size_t)(row0 + r + hb * 8) * D_ + n] = c[r];
    }
}

extern "C" void kernel_launch(void* const* d_in, const int* in_sizes, int n_in,
                              void* d_out, int out_size, void* d_ws, size_t ws_size,
                              hipStream_t stream) {
    (void)in_sizes; (void)n_in; (void)out_size; (void)ws_size;
    const float* x  = (const float*)d_in[0];
    const int*   pos = (const int*)d_in[1];
    const float* Wq = (const float*)d_in[2];
    const float* Wk = (const float*)d_in[3];
    const float* Wv = (const float*)d_in[4];
    const float* Wo = (const float*)d_in[5];
    float* outp = (float*)d_out;

    char* w = (char*)d_ws;
    const size_t SZ_X  = (size_t)NROW * D_ * 2;      // 16 MiB
    const size_t SZ_W  = (size_t)D_ * D_ * 2;        //  2 MiB
    const size_t SZ_T  = (size_t)B_ * H_ * S_ * DK_ * 2; // 16 MiB
    __bf16* xb  = (__bf16*)(w);
    __bf16* wqb = (__bf16*)(w + SZ_X);
    __bf16* wkb = (__bf16*)(w + SZ_X + SZ_W);
    __bf16* wvb = (__bf16*)(w + SZ_X + 2 * SZ_W);
    __bf16* wob = (__bf16*)(w + SZ_X + 3 * SZ_W);
    __bf16* qb  = (__bf16*)(w + SZ_X + 4 * SZ_W);
    __bf16* kb  = (__bf16*)(w + SZ_X + 4 * SZ_W + SZ_T);
    __bf16* vtb = (__bf16*)(w + SZ_X + 4 * SZ_W + 2 * SZ_T);
    __bf16* abf = (__bf16*)(w + SZ_X + 4 * SZ_W + 3 * SZ_T);

    // 1) convert inputs to bf16
    f2bf_kernel<<<(NROW * D_) / 1024, 256, 0, stream>>>(x, xb, NROW * D_);
    f2bf_kernel<<<(D_ * D_) / 1024, 256, 0, stream>>>(Wq, wqb, D_ * D_);
    f2bf_kernel<<<(D_ * D_) / 1024, 256, 0, stream>>>(Wk, wkb, D_ * D_);
    f2bf_kernel<<<(D_ * D_) / 1024, 256, 0, stream>>>(Wv, wvb, D_ * D_);
    f2bf_kernel<<<(D_ * D_) / 1024, 256, 0, stream>>>(Wo, wob, D_ * D_);

    // 2) QKV projections (+RoPE on Q/K, V stored transposed)
    dim3 gGemm(D_ / (32 * 8), NROW / 16);   // (4, 512)
    qkv_kernel<<<gGemm, 256, 0, stream>>>(xb, wqb, pos, qb,  0);
    qkv_kernel<<<gGemm, 256, 0, stream>>>(xb, wkb, pos, kb,  1);
    qkv_kernel<<<gGemm, 256, 0, stream>>>(xb, wvb, pos, vtb, 2);

    // 3) causal flash attention
    dim3 gAttn((S_ / 16) / 8, B_ * H_);     // (16, 64)
    attn_kernel<<<gAttn, 256, 0, stream>>>(qb, kb, vtb, abf);

    // 4) output projection -> fp32
    proj_kernel<<<gGemm, 256, 0, stream>>>(abf, wob, outp);
}
